// bilinear_Net_84954453115064
// MI455X (gfx1250) — matrix operand order repaired and checked by reading, not compiled, not run on hardware
//
#include <hip/hip_runtime.h>
#include <hip/hip_bf16.h>

typedef __attribute__((ext_vector_type(16))) _Float16 v16h;
typedef __attribute__((ext_vector_type(8)))  _Float16 v8h;
typedef __attribute__((ext_vector_type(8)))  float    v8f;

#define N_NODES 51200
#define NPER    100
#define NGRAPH  512
#define DEG     16
#define DIN     100
#define KPAD    128
#define BN_EPS  1e-5f

union V16H { v16h v; v8h h[2]; };

// ---------------------------------------------------------------------------
// Pack W (dout x din, row-major) as B-operand tiles for v_wmma_f32_16x16x32_f16.
// Layout: [ntile t][kstep s][lane][elem i] ; element = W^T[k][n] with
//   k = s*32 + (lane>=16 ? 8 : 0) + (i>=8 ? 16 : 0) + (i & 7) ,  n = t*16 + (lane & 15)
// zero-padded beyond (din, dout).
// ---------------------------------------------------------------------------
__global__ void pack_B(const float* __restrict__ W, _Float16* __restrict__ Bt,
                       int dout, int din, int total) {
    int idx = blockIdx.x * blockDim.x + threadIdx.x;
    if (idx >= total) return;
    int i    = idx & 15;
    int lane = (idx >> 4) & 31;
    int s    = (idx >> 9) & 3;
    int t    = idx >> 11;
    int k = s * 32 + ((lane >= 16) ? 8 : 0) + ((i >= 8) ? 16 : 0) + (i & 7);
    int n = t * 16 + (lane & 15);
    float v = 0.0f;
    if (k < din && n < dout) v = W[n * din + k];
    Bt[idx] = (_Float16)v;
}

// ---------------------------------------------------------------------------
// Layer-1 aggregation: mean over 16 in-neighbors of f32 feat -> f16 A rows
// (stride KPAD=128, zero padded for d >= 100). One block (128 thr) per node.
// ---------------------------------------------------------------------------
__global__ void agg_feat(const float* __restrict__ feat, const int* __restrict__ src,
                         _Float16* __restrict__ out) {
    __shared__ int nb[DEG];
    int n = blockIdx.x;
    int d = threadIdx.x;
    if (d < DEG) nb[d] = src[n * DEG + d];
    __syncthreads();
    float s = 0.0f;
    if (d < DIN) {
        #pragma unroll
        for (int e = 0; e < DEG; ++e) s += feat[(size_t)nb[e] * DIN + d];
        s *= (1.0f / DEG);
    }
    out[(size_t)n * KPAD + d] = (d < DIN) ? (_Float16)s : (_Float16)0.0f;
}

// Layer-2 aggregation: mean over neighbors of f16 h1 rows (stride KPAD).
__global__ void agg_h(const _Float16* __restrict__ h, const int* __restrict__ src,
                      _Float16* __restrict__ out) {
    __shared__ int nb[DEG];
    int n = blockIdx.x;
    int d = threadIdx.x;
    if (d < DEG) nb[d] = src[n * DEG + d];
    __syncthreads();
    float s = 0.0f;
    if (d < DIN) {
        #pragma unroll
        for (int e = 0; e < DEG; ++e) s += (float)h[(size_t)nb[e] * KPAD + d];
        s *= (1.0f / DEG);
    }
    out[(size_t)n * KPAD + d] = (d < DIN) ? (_Float16)s : (_Float16)0.0f;
}

// ---------------------------------------------------------------------------
// WMMA GEMM, layer 1: h1 = relu(agg1(Nx128 f16) x Wg1B + b) -> f16, stride 128.
// One wave per 16x16 output tile, K = 128 in 4 steps of 32.
// ---------------------------------------------------------------------------
__global__ void gemm1_wmma(const _Float16* __restrict__ A, const _Float16* __restrict__ Bt,
                           const float* __restrict__ bias, _Float16* __restrict__ out) {
    const int ntiles = 7;                       // 112 cols, 100 valid
    int wid  = blockIdx.x * (blockDim.x >> 5) + (threadIdx.x >> 5);
    int lane = threadIdx.x & 31;
    if (wid >= 3200 * ntiles) return;           // wave-uniform guard
    int mt = wid / ntiles, nt = wid % ntiles;
    int lr = lane & 15, khalf = lane >> 4;
    const _Float16* arow = A + (size_t)(mt * 16 + lr) * KPAD + khalf * 8;
    const _Float16* bptr = Bt + ((size_t)nt * 4) * 512 + (size_t)lane * 16;
    v8f c = {};
    #pragma unroll
    for (int s = 0; s < 4; ++s) {
        V16H a, b;
        a.h[0] = *(const v8h*)(arow + s * 32);
        a.h[1] = *(const v8h*)(arow + s * 32 + 16);
        b.v    = *(const v16h*)(bptr + s * 512);
        c = __builtin_amdgcn_wmma_f32_16x16x32_f16(false, a.v, false, b.v,
                                                   (short)0, c, false, false);
    }
    int ncol = nt * 16 + lr;
    if (ncol < DIN) {
        float bb = bias[ncol];
        int mbase = mt * 16 + 8 * khalf;        // C layout: VGPR r -> M = r + 8*khalf
        #pragma unroll
        for (int r = 0; r < 8; ++r) {
            float v = c[r] + bb;
            v = v > 0.0f ? v : 0.0f;
            out[(size_t)(mbase + r) * KPAD + ncol] = (_Float16)v;
        }
    }
}

// WMMA GEMM, layer 2: h2 = relu(agg2(Nx128 f16) x Wg2B + b) -> f32, stride 32.
__global__ void gemm2_wmma(const _Float16* __restrict__ A, const _Float16* __restrict__ Bt,
                           const float* __restrict__ bias, float* __restrict__ out) {
    const int ntiles = 2;                       // 32 cols, 20 valid
    int wid  = blockIdx.x * (blockDim.x >> 5) + (threadIdx.x >> 5);
    int lane = threadIdx.x & 31;
    if (wid >= 3200 * ntiles) return;
    int mt = wid / ntiles, nt = wid % ntiles;
    int lr = lane & 15, khalf = lane >> 4;
    const _Float16* arow = A + (size_t)(mt * 16 + lr) * KPAD + khalf * 8;
    const _Float16* bptr = Bt + ((size_t)nt * 4) * 512 + (size_t)lane * 16;
    v8f c = {};
    #pragma unroll
    for (int s = 0; s < 4; ++s) {
        V16H a, b;
        a.h[0] = *(const v8h*)(arow + s * 32);
        a.h[1] = *(const v8h*)(arow + s * 32 + 16);
        b.v    = *(const v16h*)(bptr + s * 512);
        c = __builtin_amdgcn_wmma_f32_16x16x32_f16(false, a.v, false, b.v,
                                                   (short)0, c, false, false);
    }
    int ncol = nt * 16 + lr;
    if (ncol < 20) {
        float bb = bias[ncol];
        int mbase = mt * 16 + 8 * khalf;
        #pragma unroll
        for (int r = 0; r < 8; ++r) {
            float v = c[r] + bb;
            v = v > 0.0f ? v : 0.0f;
            out[(size_t)(mbase + r) * 32 + ncol] = v;
        }
    }
}

// ---------------------------------------------------------------------------
// Per-graph mean pooling over 100 nodes + FiLM: hg = pool*sigmoid(sf@Wg+bg) + (sf@Wb+bb)
// One block (32 thr) per graph; hg stored stride 32 (20 valid).
// ---------------------------------------------------------------------------
__global__ void pool_film(const float* __restrict__ h2, const float* __restrict__ self_feat,
                          const float* __restrict__ Wg, const float* __restrict__ bg,
                          const float* __restrict__ Wb, const float* __restrict__ bb,
                          float* __restrict__ hg) {
    int g = blockIdx.x;
    int d = threadIdx.x;
    float o = 0.0f;
    if (d < 20) {
        float pool = 0.0f;
        const float* base = h2 + (size_t)g * NPER * 32 + d;
        for (int i = 0; i < NPER; ++i) pool += base[(size_t)i * 32];
        pool *= (1.0f / NPER);
        const float* sf = self_feat + (size_t)g * 32;
        float ga = bg[d], be = bb[d];
        #pragma unroll
        for (int k = 0; k < 32; ++k) {
            ga += sf[k] * Wg[d * 32 + k];
            be += sf[k] * Wb[d * 32 + k];
        }
        ga = 1.0f / (1.0f + __expf(-ga));
        o = pool * ga + be;
    }
    hg[(size_t)g * 32 + d] = o;
}

// Dense fully-connected: Y[g][j] = X[g]·W[j] + b[j]
__global__ void fc(const float* __restrict__ X, const float* __restrict__ W,
                   const float* __restrict__ b, float* __restrict__ Y,
                   int dout, int din, int xstride) {
    int idx = blockIdx.x * blockDim.x + threadIdx.x;
    if (idx >= NGRAPH * dout) return;
    int g = idx / dout, j = idx % dout;
    float s = b[j];
    const float* x = X + (size_t)g * xstride;
    const float* w = W + (size_t)j * din;
    for (int k = 0; k < din; ++k) s += x[k] * w[k];
    Y[(size_t)g * dout + j] = s;
}

// Batch-norm (biased var over 512 rows) + ReLU, in place. One block per column.
__global__ void bn_relu(float* __restrict__ X, const float* __restrict__ g,
                        const float* __restrict__ b, int cols) {
    __shared__ float ssum[256], ssq[256];
    int j = blockIdx.x;
    int tid = threadIdx.x;
    float s = 0.0f, q = 0.0f;
    for (int r = tid; r < NGRAPH; r += 256) {
        float v = X[(size_t)r * cols + j];
        s += v; q += v * v;
    }
    ssum[tid] = s; ssq[tid] = q;
    __syncthreads();
    for (int off = 128; off > 0; off >>= 1) {
        if (tid < off) { ssum[tid] += ssum[tid + off]; ssq[tid] += ssq[tid + off]; }
        __syncthreads();
    }
    float mu  = ssum[0] * (1.0f / NGRAPH);
    float var = ssq[0] * (1.0f / NGRAPH) - mu * mu;
    float inv = rsqrtf(var + BN_EPS);
    float gg = g[j], bb = b[j];
    for (int r = tid; r < NGRAPH; r += 256) {
        size_t i = (size_t)r * cols + j;
        float v = gg * (X[i] - mu) * inv + bb;
        X[i] = v > 0.0f ? v : 0.0f;
    }
}

extern "C" void kernel_launch(void* const* d_in, const int* in_sizes, int n_in,
                              void* d_out, int out_size, void* d_ws, size_t ws_size,
                              hipStream_t stream) {
    const float* feat      = (const float*)d_in[0];
    const float* self_feat = (const float*)d_in[1];
    const float* W_gc1 = (const float*)d_in[2];  const float* b_gc1 = (const float*)d_in[3];
    const float* W_gc2 = (const float*)d_in[4];  const float* b_gc2 = (const float*)d_in[5];
    const float* W_fc1 = (const float*)d_in[6];  const float* b_fc1 = (const float*)d_in[7];
    const float* W_fc2 = (const float*)d_in[8];  const float* b_fc2 = (const float*)d_in[9];
    const float* W_fc3 = (const float*)d_in[10]; const float* b_fc3 = (const float*)d_in[11];
    const float* bn1_g = (const float*)d_in[12]; const float* bn1_b = (const float*)d_in[13];
    const float* bn2_g = (const float*)d_in[14]; const float* bn2_b = (const float*)d_in[15];
    const float* W_gam = (const float*)d_in[16]; const float* b_gam = (const float*)d_in[17];
    const float* W_bet = (const float*)d_in[18]; const float* b_bet = (const float*)d_in[19];
    const int*   src   = (const int*)d_in[20];
    float* out = (float*)d_out;

    char* ws = (char*)d_ws;
    _Float16* Wg1B = (_Float16*)(ws + 0);          //  7*2048 f16 = 28 KB
    _Float16* Wg2B = (_Float16*)(ws + 32768);      //  2*2048 f16
    _Float16* agg1 = (_Float16*)(ws + 65536);      //  N*128 f16 = 12.5 MB
    _Float16* h1   = (_Float16*)(ws + 13172736);   //  N*128 f16
    _Float16* agg2 = (_Float16*)(ws + 26279936);   //  N*128 f16
    float*    h2   = (float*)   (ws + 39387136);   //  N*32 f32 = 6.25 MB
    float*    hg   = (float*)   (ws + 45940736);   //  512*32 f32
    float*    Y1   = (float*)   (ws + 46006272);   //  512*128 f32
    float*    Y2   = (float*)   (ws + 46268416);   //  512*32 f32

    pack_B<<<(14336 + 255) / 256, 256, 0, stream>>>(W_gc1, Wg1B, 100, 100, 14336);
    pack_B<<<(4096  + 255) / 256, 256, 0, stream>>>(W_gc2, Wg2B, 20, 100, 4096);

    agg_feat<<<N_NODES, 128, 0, stream>>>(feat, src, agg1);
    gemm1_wmma<<<2800, 256, 0, stream>>>(agg1, Wg1B, b_gc1, h1);   // 3200*7 waves
    agg_h<<<N_NODES, 128, 0, stream>>>(h1, src, agg2);
    gemm2_wmma<<<800, 256, 0, stream>>>(agg2, Wg2B, b_gc2, h2);    // 3200*2 waves

    pool_film<<<NGRAPH, 32, 0, stream>>>(h2, self_feat, W_gam, b_gam, W_bet, b_bet, hg);

    fc<<<(NGRAPH * 128 + 255) / 256, 256, 0, stream>>>(hg, W_fc1, b_fc1, Y1, 128, 20, 32);
    bn_relu<<<128, 256, 0, stream>>>(Y1, bn1_g, bn1_b, 128);
    fc<<<(NGRAPH * 32 + 255) / 256, 256, 0, stream>>>(Y1, W_fc2, b_fc2, Y2, 32, 128, 128);
    bn_relu<<<32, 256, 0, stream>>>(Y2, bn2_g, bn2_b, 32);
    fc<<<(NGRAPH + 255) / 256, 256, 0, stream>>>(Y2, W_fc3, b_fc3, out, 1, 32, 32);
}
